// Glimpse_3908420239956
// MI455X (gfx1250) — matrix operand order repaired and checked by reading, not compiled, not run on hardware
//
#include <hip/hip_runtime.h>

// ---------------- problem constants ----------------
#define B_SZ   32
#define NTOT   10000
#define DIM    256
#define NH     16
#define CHUNK  384              // rows of node_emb per workgroup (12 K-tiles of 32)
#define NC     27               // ceil(NTOT / CHUNK)
#define RT     (CHUNK/16)       // 24 row tiles -> exactly 3 per wave
#define KT     (CHUNK/32)       // 12 K-steps for the w^T @ X GEMM
#define NEGINF (-3.402823466e38f)

typedef __attribute__((ext_vector_type(16))) __bf16 v16bf;
typedef __attribute__((ext_vector_type(8)))  __bf16 v8bf;
typedef __attribute__((ext_vector_type(4)))  __bf16 v4bf;
typedef __attribute__((ext_vector_type(8)))  float  v8f;

__device__ __forceinline__ v8f wmma_bf16(v16bf a, v16bf b, v8f c) {
  // D = A(16x32 bf16) * B(32x16 bf16) + C(16x16 f32)
  return __builtin_amdgcn_wmma_f32_16x16x32_bf16(false, a, false, b, (short)0, c,
                                                 false, false);
}

// LDS layout offsets (bytes) inside dynamic smem
#define OFF_X     0
#define OFF_SLOG  (CHUNK * DIM * 2)                 // 196608
#define OFF_WT    (OFF_SLOG + CHUNK * NH * 4)       // 221184
#define OFF_FB    (OFF_WT + CHUNK * NH * 2)         // 233472
#define OFF_RED   (OFF_FB + CHUNK * 4)              // 235008
#define OFF_MS    (OFF_RED + 256 * 4)               // 236032
#define OFF_SS    (OFF_MS + NH * 4)                 // 236096
#define SMEM_SZ   (OFF_SS + NH * 4)                 // 236160

// ---------------------------------------------------------------------------
// Kernel 1: q = ctx @ Wq^T ; QKt[b][i][h] = sum_j q[b,h*16+j] * Wk[h*16+j, i]
// (folds the query into Wk so K is never materialized). Output bf16, [b][d][h]
// layout = exactly the WMMA B-matrix feed order for the logits GEMM.
// ---------------------------------------------------------------------------
__global__ __launch_bounds__(256)
void glimpse_prep(const float* __restrict__ ctx, const float* __restrict__ Wq,
                  const float* __restrict__ Wk, __bf16* __restrict__ QKt) {
  __shared__ float cx[DIM];
  __shared__ float q[DIM];
  const int b = blockIdx.x, t = threadIdx.x;
  cx[t] = ctx[(size_t)b * DIM + t];
  __syncthreads();
  {
    float a = 0.f;
    const float* wrow = Wq + (size_t)t * DIM;
    #pragma unroll 8
    for (int i = 0; i < DIM; ++i) a += wrow[i] * cx[i];
    q[t] = a;
  }
  __syncthreads();
  const int h  = t & 15;
  const int i0 = t >> 4;
  #pragma unroll
  for (int k = 0; k < 16; ++k) {
    const int i = i0 + 16 * k;
    float s = 0.f;
    #pragma unroll
    for (int j = 0; j < 16; ++j)
      s += q[h * 16 + j] * Wk[(size_t)(h * 16 + j) * DIM + i];
    QKt[((size_t)b * DIM + i) * NH + h] = (__bf16)s;
  }
}

// ---------------------------------------------------------------------------
// Kernel 2: fused single pass over node_emb.
//  - stage a 384x256 chunk as bf16 in LDS (192 KB) + mask-bias array
//  - GEMM1 (WMMA): logits = X_chunk @ QKt, 3 row tiles per wave interleaved
//  - split-softmax over the chunk; weights stored bf16 head-major (WMMA A feed)
//  - GEMM2 (WMMA): xacc_p = w^T @ X_chunk, 2 d-tiles per wave interleaved
// grid = (32, 27), 256 threads = 8 waves, 1 WG per WGP (LDS-bound).
// ---------------------------------------------------------------------------
__global__ __launch_bounds__(256)
void glimpse_attn(const float* __restrict__ node,
                  const unsigned char* __restrict__ feas,
                  const __bf16* __restrict__ QKt,
                  float* __restrict__ mP, float* __restrict__ sP,
                  float* __restrict__ xaccP) {
  extern __shared__ char smem[];
  __bf16* ldsX = (__bf16*)(smem + OFF_X);    // [CHUNK][DIM] bf16
  float*  Slog = (float*) (smem + OFF_SLOG); // [CHUNK][NH] logits f32
  __bf16* Wt   = (__bf16*)(smem + OFF_WT);   // [NH][CHUNK] exp-weights bf16
  float*  fb   = (float*) (smem + OFF_FB);   // [CHUNK] mask bias (0 or -inf)
  float*  red  = (float*) (smem + OFF_RED);  // [16][NH] reduction scratch
  float*  mS   = (float*) (smem + OFF_MS);   // [NH]
  float*  sS   = (float*) (smem + OFF_SS);   // [NH]

  const int b = blockIdx.x, chunk = blockIdx.y;
  const int base = chunk * CHUNK;
  const int tid  = threadIdx.x;
  const int lane = tid & 31, wave = tid >> 5;
  const int lo = lane & 15, hi = lane >> 4;

  // ---- Phase 0a: mask bias for the chunk (branch-free hot loop later) ----
  for (int r = tid; r < CHUNK; r += 256) {
    const int gr = base + r;
    const bool ok = (gr < NTOT) && (feas[(size_t)b * NTOT + gr] != 0);
    fb[r] = ok ? 0.f : NEGINF;
  }
  // ---- Phase 0b: global fp32 -> LDS bf16 (zero-fill rows past NTOT) ----
  for (int idx = tid; idx < CHUNK * (DIM / 4); idx += 256) {
    const int r = idx >> 6, c4 = idx & 63;
    const int gr = base + r;
    v4bf o;
    if (gr < NTOT) {
      const float4 v = *(const float4*)(node + ((size_t)b * NTOT + gr) * DIM + c4 * 4);
      o[0] = (__bf16)v.x; o[1] = (__bf16)v.y; o[2] = (__bf16)v.z; o[3] = (__bf16)v.w;
    } else {
      o[0] = o[1] = o[2] = o[3] = (__bf16)0.f;
    }
    *(v4bf*)(ldsX + (size_t)r * DIM + c4 * 4) = o;
  }

  // ---- Phase 1: QKt B-fragments, register-resident for the whole chunk.
  // B 32x16 layout: element j of lane -> k = 16*hi + j, column n(head) = lo.
  v16bf Bq[8];
  #pragma unroll
  for (int s = 0; s < 8; ++s)
    #pragma unroll
    for (int j = 0; j < 16; ++j)
      Bq[s][j] = QKt[((size_t)b * DIM + (32 * s + hi * 16 + j)) * NH + lo];
  __syncthreads();

  // ---- Phase 2: logits via WMMA; 3 independent accumulators per wave ----
  {
    v8f c0 = {}, c1 = {}, c2 = {};
    const __bf16* r0 = ldsX + (size_t)((wave      ) * 16 + lo) * DIM;
    const __bf16* r1 = ldsX + (size_t)((wave +  8 ) * 16 + lo) * DIM;
    const __bf16* r2 = ldsX + (size_t)((wave + 16 ) * 16 + lo) * DIM;
    #pragma unroll
    for (int s = 0; s < 8; ++s) {                  // K = 256 in 8 steps
      v16bf a0, a1, a2;  // A 16x32: one contiguous 16-elem run per lane, k = 16*hi + j
      a0 = *(const v16bf*)(r0 + 32 * s + hi * 16);
      a1 = *(const v16bf*)(r1 + 32 * s + hi * 16);
      a2 = *(const v16bf*)(r2 + 32 * s + hi * 16);
      c0 = wmma_bf16(a0, Bq[s], c0);
      c1 = wmma_bf16(a1, Bq[s], c1);
      c2 = wmma_bf16(a2, Bq[s], c2);
    }
    const v8f cc[3] = {c0, c1, c2};
    #pragma unroll
    for (int i = 0; i < 3; ++i) {
      const int t = wave + 8 * i;
      #pragma unroll
      for (int r = 0; r < 8; ++r) {          // D: m = r + 8*hi, head = lo
        const int row = t * 16 + r + hi * 8;
        Slog[row * NH + lo] = cc[i][r] * 0.25f + fb[row];  // scale + mask bias
      }
    }
  }
  __syncthreads();

  // ---- Phase 3: per-head chunk max (16 partials per head) ----
  const int h = tid & 15, part = tid >> 4;
  float lm = NEGINF;
  for (int r = part; r < CHUNK; r += 16) lm = fmaxf(lm, Slog[r * NH + h]);
  red[part * 16 + h] = lm;
  __syncthreads();
  if (tid < NH) {
    float mm = NEGINF;
    #pragma unroll
    for (int p = 0; p < 16; ++p) mm = fmaxf(mm, red[p * 16 + tid]);
    mS[tid] = mm;
  }
  __syncthreads();

  // ---- Phase 4: exp weights -> bf16 head-major (WMMA A layout) + sums ----
  const float mh = mS[h];
  float ls = 0.f;
  for (int r = part; r < CHUNK; r += 16) {
    const float w = __expf(Slog[r * NH + h] - mh);
    Wt[(size_t)h * CHUNK + r] = (__bf16)w;
    ls += w;
  }
  red[part * 16 + h] = ls;
  __syncthreads();
  if (tid < NH) {
    float ss = 0.f;
    #pragma unroll
    for (int p = 0; p < 16; ++p) ss += red[p * 16 + tid];
    sS[tid] = ss;
  }
  __syncthreads();

  // ---- Phase 5: xacc_p = w^T @ X (M=16 heads, N=16 d, K=384);
  //      2 d-tiles per wave, shared A-fragment, interleaved WMMAs ----
  float* dst = xaccP + (size_t)(b * NC + chunk) * NH * DIM;
  {
    const int d0 = wave * 16, d1 = (wave + 8) * 16;
    v8f acc0 = {}, acc1 = {};
    const __bf16* wrow = Wt + (size_t)lo * CHUNK;   // A row m = head = lo
    for (int kt = 0; kt < KT; ++kt) {
      const int nb = kt * 32;
      v16bf a;     // same contiguous-run layout as GEMM1 A: k = 16*hi + j
      a = *(const v16bf*)(wrow + nb + hi * 16);
      v16bf b0, b1;  // B: k = 16*hi + j -> row nb+16*hi+j, n = lo -> d-col
      #pragma unroll
      for (int j = 0; j < 16; ++j) {
        const __bf16* xr = ldsX + (size_t)(nb + hi * 16 + j) * DIM + lo;
        b0[j] = xr[d0];
        b1[j] = xr[d1];
      }
      acc0 = wmma_bf16(a, b0, acc0);
      acc1 = wmma_bf16(a, b1, acc1);
    }
    #pragma unroll
    for (int r = 0; r < 8; ++r) {            // D: head = r + 8*hi, d = dX + lo
      dst[(size_t)(r + hi * 8) * DIM + d0 + lo] = acc0[r];
      dst[(size_t)(r + hi * 8) * DIM + d1 + lo] = acc1[r];
    }
  }
  if (tid < NH) {
    mP[(size_t)(b * NC + chunk) * NH + tid] = mS[tid];
    sP[(size_t)(b * NC + chunk) * NH + tid] = sS[tid];
  }
}

// ---------------------------------------------------------------------------
// Kernel 3: merge chunk partials (flash combine), then head_ctx = Wv_h @ xattn_h
// and out = head_ctx @ Wo^T. One WG per batch element; everything is tiny.
// ---------------------------------------------------------------------------
__global__ __launch_bounds__(256)
void glimpse_merge(const float* __restrict__ mP, const float* __restrict__ sP,
                   const float* __restrict__ xaccP,
                   const float* __restrict__ Wv, const float* __restrict__ Wo,
                   float* __restrict__ out) {
  __shared__ float xattn[NH * DIM];
  __shared__ float hc[DIM];
  __shared__ float mH[NH], sH[NH];
  const int b = blockIdx.x, tid = threadIdx.x;

  if (tid < NH) {
    float m = NEGINF;
    for (int c = 0; c < NC; ++c) m = fmaxf(m, mP[(size_t)(b * NC + c) * NH + tid]);
    float s = 0.f;
    for (int c = 0; c < NC; ++c)
      s += sP[(size_t)(b * NC + c) * NH + tid] *
           __expf(mP[(size_t)(b * NC + c) * NH + tid] - m);
    mH[tid] = m; sH[tid] = s;
  }
  __syncthreads();

  const int h = tid >> 4, d0 = tid & 15;
  float acc[16];
  #pragma unroll
  for (int k = 0; k < 16; ++k) acc[k] = 0.f;
  for (int c = 0; c < NC; ++c) {
    const float f = __expf(mP[(size_t)(b * NC + c) * NH + h] - mH[h]);
    if (f > 0.f) {
      const float* src = xaccP + ((size_t)(b * NC + c) * NH + h) * DIM;
      #pragma unroll
      for (int k = 0; k < 16; ++k) acc[k] += f * src[d0 + 16 * k];
    }
  }
  const float inv = 1.f / sH[h];
  #pragma unroll
  for (int k = 0; k < 16; ++k) xattn[h * DIM + d0 + 16 * k] = acc[k] * inv;
  __syncthreads();

  {   // head_ctx[b, tid] (tid = h*16 + d') = xattn[h,:] . Wv[tid,:]
    float a = 0.f;
    const float* wrow = Wv + (size_t)tid * DIM;
    const float* xr   = xattn + (size_t)(tid >> 4) * DIM;
    #pragma unroll 8
    for (int i = 0; i < DIM; ++i) a += wrow[i] * xr[i];
    hc[tid] = a;
  }
  __syncthreads();
  {   // out = head_ctx @ Wo^T
    float a = 0.f;
    const float* wrow = Wo + (size_t)tid * DIM;
    #pragma unroll 8
    for (int j = 0; j < DIM; ++j) a += wrow[j] * hc[j];
    out[(size_t)b * DIM + tid] = a;
  }
}

// ---------------------------------------------------------------------------
extern "C" void kernel_launch(void* const* d_in, const int* in_sizes, int n_in,
                              void* d_out, int out_size, void* d_ws, size_t ws_size,
                              hipStream_t stream) {
  (void)in_sizes; (void)n_in; (void)out_size; (void)ws_size;
  const float*         ctx  = (const float*)d_in[0];
  const float*         node = (const float*)d_in[1];
  const unsigned char* feas = (const unsigned char*)d_in[2];   // jnp.bool_ = 1 byte
  const float*         Wq   = (const float*)d_in[3];
  const float*         Wk   = (const float*)d_in[4];
  const float*         Wv   = (const float*)d_in[5];
  const float*         Wo   = (const float*)d_in[6];
  float* out = (float*)d_out;

  char* ws = (char*)d_ws;
  size_t off = 0;
  __bf16* QKt = (__bf16*)(ws + off); off += (size_t)B_SZ * DIM * NH * 2;  // 256 KiB
  float* mP    = (float*)(ws + off); off += (size_t)B_SZ * NC * NH * 4;
  float* sP    = (float*)(ws + off); off += (size_t)B_SZ * NC * NH * 4;
  float* xaccP = (float*)(ws + off);                                      // ~14.2 MiB

  glimpse_prep<<<B_SZ, 256, 0, stream>>>(ctx, Wq, Wk, QKt);
  glimpse_attn<<<dim3(B_SZ, NC), 256, SMEM_SZ, stream>>>(node, feas, QKt,
                                                         mP, sP, xaccP);
  glimpse_merge<<<B_SZ, 256, 0, stream>>>(mP, sP, xaccP, Wv, Wo, out);
}
